// MultiHeadSelfAttention_7756710936699
// MI455X (gfx1250) — matrix operand order repaired and checked by reading, not compiled, not run on hardware
//
#include <hip/hip_runtime.h>
#include <hip/hip_bf16.h>
#include <stdint.h>

#define B_  2
#define S_  2048
#define D_  1024
#define H_  16
#define DH_ 64
#define M_  (B_*S_)          // 4096 rows total

typedef unsigned short u16;
typedef __attribute__((ext_vector_type(16))) __bf16 v16bf;
typedef __attribute__((ext_vector_type(8)))  float  v8f;
typedef __attribute__((ext_vector_type(8)))  u16    us8;
typedef __attribute__((ext_vector_type(16))) u16    us16;

// Scheduling pins: 0x0020 = VMEM read group, 0x0008 = MFMA/WMMA group.
#if __has_builtin(__builtin_amdgcn_sched_group_barrier)
#define SCHED_VMEM(n) __builtin_amdgcn_sched_group_barrier(0x0020, (n), 0)
#define SCHED_WMMA(n) __builtin_amdgcn_sched_group_barrier(0x0008, (n), 0)
#else
#define SCHED_VMEM(n)
#define SCHED_WMMA(n)
#endif
#if __has_builtin(__builtin_amdgcn_sched_barrier)
#define SCHED_FENCE() __builtin_amdgcn_sched_barrier(0)
#else
#define SCHED_FENCE()
#endif

union BF16x16 { us16 u; v16bf b; };

static __device__ inline u16 f2bf(float f) {              // RNE fp32 -> bf16
  unsigned x = __float_as_uint(f);
  unsigned r = (x + 0x7FFFu + ((x >> 16) & 1u)) >> 16;
  return (u16)r;
}

static __device__ inline v8f vzero8() {
  v8f z;
#pragma unroll
  for (int i = 0; i < 8; ++i) z[i] = 0.f;
  return z;
}

// A-operand (16x32 bf16, MxK): lane L holds row (L&15);
// elements 0..15 = K[kk + 16*half .. +15] (contiguous)
static __device__ inline v16bf make_a(const u16* row, int kk, int half) {
  us8 lo = *(const us8*)(row + kk + half * 16);
  us8 hi = *(const us8*)(row + kk + half * 16 + 8);
  BF16x16 t;
#pragma unroll
  for (int i = 0; i < 8; ++i) { t.u[i] = lo[i]; t.u[i + 8] = hi[i]; }
  return t.b;
}

// B-operand (32x16 bf16, KxN): lane L = column (L&15); 16 contiguous K values
// starting at K = 16*half. Caller passes p already offset to that K.
static __device__ inline v16bf ldb16(const u16* p) {
  BF16x16 t;
  t.u = *(const us16*)p;   // 32B contiguous (two global_load_b128)
  return t.b;
}

static __device__ inline v8f wmma_bf16(v16bf a, v16bf b, v8f c) {
  return __builtin_amdgcn_wmma_f32_16x16x32_bf16(false, a, false, b,
                                                 (short)0, c, false, false);
}

// ---------------- fp32 -> bf16 convert ----------------
__global__ void k_to_bf16(const float* __restrict__ in, u16* __restrict__ out, int n) {
  int i = blockIdx.x * blockDim.x + threadIdx.x;
  int stride = gridDim.x * blockDim.x;
  for (; i < n; i += stride) out[i] = f2bf(in[i]);
}

// ---------------- QKV projection: out = x @ W.T ----------------
// One wave computes a 16x64 block. K loop is fully unrolled (32 chunks) with a
// mod-3 operand pipeline at prefetch distance 2; sched_group_barriers pin the
// VMEMx10 / WMMAx4 interleave so ~2 chunks of loads stay in flight.
// transposed==0 -> out[b,h,s,dh] (Q,K); 1 -> out[b,h,dh,s] (V^T).
__global__ __launch_bounds__(128)
void k_proj(const u16* __restrict__ xb, const u16* __restrict__ wb,
            u16* __restrict__ out, int transposed) {
  const int lane = threadIdx.x & 31, lr = lane & 15, half = lane >> 4;
  const int wid = blockIdx.x * (blockDim.x >> 5) + (threadIdx.x >> 5);
  const int nb = wid & 15;            // head (64-wide N block)
  const int mt = wid >> 4;            // 0..255 M tiles
  const u16* arow = xb + (size_t)(mt * 16 + lr) * D_;
  const u16* wrow[4];
#pragma unroll
  for (int t = 0; t < 4; ++t)
    wrow[t] = wb + (size_t)(nb * 64 + t * 16 + lr) * D_ + half * 16;
  v8f acc[4] = { vzero8(), vzero8(), vzero8(), vzero8() };

  constexpr int NC = D_ / 32;         // 32 K-chunks
  v16bf Ap[3];
  v16bf Bp[3][4];
#pragma unroll
  for (int c = 0; c < 2; ++c) {
    Ap[c] = make_a(arow, c * 32, half);
#pragma unroll
    for (int t = 0; t < 4; ++t) Bp[c][t] = ldb16(wrow[t] + c * 32);
  }
#pragma unroll
  for (int c = 0; c < NC; ++c) {
    if (c + 2 < NC) {                 // folds at compile time (full unroll)
      Ap[(c + 2) % 3] = make_a(arow, (c + 2) * 32, half);
#pragma unroll
      for (int t = 0; t < 4; ++t) Bp[(c + 2) % 3][t] = ldb16(wrow[t] + (c + 2) * 32);
    }
#pragma unroll
    for (int t = 0; t < 4; ++t) acc[t] = wmma_bf16(Ap[c % 3], Bp[c % 3][t], acc[t]);
  }
  // Pin the pipeline: 2 chunks of loads up front, then [WMMAx4, VMEMx10]*30,
  // then the two drained chunks.
  SCHED_VMEM(20);
#pragma unroll
  for (int c = 0; c < NC - 2; ++c) { SCHED_WMMA(4); SCHED_VMEM(10); }
  SCHED_WMMA(4);
  SCHED_WMMA(4);
  SCHED_FENCE();                      // keep the store epilogue out of the pipe

#pragma unroll
  for (int t = 0; t < 4; ++t)
#pragma unroll
    for (int r = 0; r < 8; ++r) {
      int m = mt * 16 + r + half * 8;
      int b = m >> 11, s = m & (S_ - 1);
      int n = nb * 64 + t * 16 + lr;
      int h = n >> 6, dh = n & 63;
      size_t idx = transposed
          ? ((size_t)((b * H_ + h) * DH_ + dh)) * S_ + s
          : ((size_t)((b * H_ + h) * S_ + s)) * DH_ + dh;
      out[idx] = f2bf(acc[t][r]);
    }
}

// ---------------- column softmax stats (softmax over QUERY axis) -----------
// One wave owns 16 key-columns of one (b,h); streams all q tiles via WMMA,
// online max/sum over the query axis. B operands are loop-invariant. Q-tile A
// operands use a mod-4 slot pipeline (prefetch distance 2) inside an
// unroll-16 body; sched_group_barriers pin the WMMAx2 / VMEMx4 interleave.
__global__ __launch_bounds__(128)
void k_colstats(const u16* __restrict__ qh, const u16* __restrict__ kh,
                float* __restrict__ colM, float* __restrict__ colRZ) {
  const int lane = threadIdx.x & 31, lr = lane & 15, half = lane >> 4;
  const int wid = blockIdx.x * (blockDim.x >> 5) + (threadIdx.x >> 5);
  const int kt = wid & 127;           // 16-col block within S
  const int bh = wid >> 7;            // 0..31
  const u16* kbase = kh + ((size_t)bh * S_ + kt * 16 + lr) * DH_;
  v16bf B0 = ldb16(kbase + 0 + half * 16);     // K-rows, d = 0..31
  v16bf B1 = ldb16(kbase + 32 + half * 16);    // d = 32..63
  const u16* qb = qh + (size_t)bh * S_ * DH_;

  float runM = -1e30f, runS = 0.f;
  auto step = [&](const v8f& s) {
    float sv[8], pm = -1e30f;
#pragma unroll
    for (int r = 0; r < 8; ++r) { sv[r] = s[r] * 0.125f; pm = fmaxf(pm, sv[r]); }
    float ps = 0.f;
#pragma unroll
    for (int r = 0; r < 8; ++r) ps += __expf(sv[r] - pm);
    float pm2 = __shfl_xor(pm, 16, 32);          // combine row halves
    float ps2 = __shfl_xor(ps, 16, 32);
    float tm = fmaxf(pm, pm2);
    float ts = ps * __expf(pm - tm) + ps2 * __expf(pm2 - tm);
    float nm = fmaxf(runM, tm);
    runS = runS * __expf(runM - nm) + ts * __expf(tm - nm);
    runM = nm;
  };
  auto qrow = [&](int qt) { return qb + (size_t)(qt * 16 + lr) * DH_; };

  v16bf PA0[4], PA1[4];
#pragma unroll
  for (int c = 0; c < 2; ++c) {
    PA0[c] = make_a(qrow(c), 0, half);
    PA1[c] = make_a(qrow(c), 32, half);
  }
  for (int g = 0; g < 7; ++g) {                 // groups 0..6: prefetch always valid
#pragma unroll
    for (int i = 0; i < 16; ++i) {
      const int qt = g * 16 + i;
      const int sp = (i + 2) & 3;               // compile-time slot
      PA0[sp] = make_a(qrow(qt + 2), 0, half);
      PA1[sp] = make_a(qrow(qt + 2), 32, half);
      v8f s = vzero8();
      s = wmma_bf16(PA0[i & 3], B0, s);
      s = wmma_bf16(PA1[i & 3], B1, s);
      step(s);
    }
#pragma unroll
    for (int i = 0; i < 16; ++i) { SCHED_WMMA(2); SCHED_VMEM(4); }
  }
#pragma unroll
  for (int i = 0; i < 16; ++i) {                // last group, guard folds
    const int qt = 112 + i;
    if (i + 2 < 16) {
      const int sp = (i + 2) & 3;
      PA0[sp] = make_a(qrow(qt + 2), 0, half);
      PA1[sp] = make_a(qrow(qt + 2), 32, half);
    }
    v8f s = vzero8();
    s = wmma_bf16(PA0[i & 3], B0, s);
    s = wmma_bf16(PA1[i & 3], B1, s);
    step(s);
  }
  if (half == 0) {
    size_t idx = (size_t)bh * S_ + kt * 16 + lr;
    colM[idx]  = runM;
    colRZ[idx] = 1.0f / runS;
  }
}

// ---------------- ctx = P @ V' (flash-style recompute) ----------------
// One wave owns a 16x64 ctx block of one (b,h,q-tile). Per 32-key chunk all 8
// B operands (4 K-row tiles + 4 V^T tiles) are issued as one batch up front so
// the V^T loads fly during the score WMMAs, exp VALU work and the LDS lane-
// transpose; P is re-fed to WMMA as A-operand via a private 1KB LDS slab.
__global__ __launch_bounds__(128)
void k_ctx(const u16* __restrict__ qh, const u16* __restrict__ kh,
           const u16* __restrict__ vT, const float* __restrict__ colM,
           const float* __restrict__ colRZ, u16* __restrict__ ctxb) {
  __shared__ __align__(16) u16 plds[4][16 * 32];     // per-wave P staging
  const int lane = threadIdx.x & 31, lr = lane & 15, half = lane >> 4;
  const int w = (threadIdx.x >> 5) & 3;
  const int wid = blockIdx.x * (blockDim.x >> 5) + (threadIdx.x >> 5);
  const int qt = wid & 127;
  const int bh = wid >> 7;
  const int b = bh / H_, h = bh % H_;

  const u16* arow = qh + ((size_t)bh * S_ + qt * 16 + lr) * DH_;
  v16bf AQ0 = make_a(arow, 0, half);
  v16bf AQ1 = make_a(arow, 32, half);
  v8f acc[4] = { vzero8(), vzero8(), vzero8(), vzero8() };
  const float* mptr = colM + (size_t)bh * S_;
  const float* zptr = colRZ + (size_t)bh * S_;
  const u16* vbase = vT + (size_t)bh * DH_ * S_;
  u16* myl = plds[w];

  for (int kb = 0; kb < S_; kb += 32) {
    // ---- batch-issue all global loads for this iteration ----
    v16bf Bk[2][2], Bv[4];
#pragma unroll
    for (int ti = 0; ti < 2; ++ti) {
      const u16* krow = kh + ((size_t)bh * S_ + kb + ti * 16 + lr) * DH_;
      Bk[ti][0] = ldb16(krow + half * 16);
      Bk[ti][1] = ldb16(krow + 32 + half * 16);
    }
#pragma unroll
    for (int t = 0; t < 4; ++t)
      Bv[t] = ldb16(vbase + (size_t)(t * 16 + lr) * S_ + kb + half * 16);

    // ---- score tiles -> exp -> LDS (C-layout) ----
#pragma unroll
    for (int ti = 0; ti < 2; ++ti) {
      v8f s = vzero8();
      s = wmma_bf16(AQ0, Bk[ti][0], s);
      s = wmma_bf16(AQ1, Bk[ti][1], s);
      int col = kb + ti * 16 + lr;
      float mc = mptr[col], rz = zptr[col];
#pragma unroll
      for (int r = 0; r < 8; ++r) {
        float p = __expf(s[r] * 0.125f - mc) * rz;
        myl[(r + half * 8) * 32 + ti * 16 + lr] = f2bf(p);
      }
    }
    asm volatile("s_wait_dscnt 0" ::: "memory");    // P tile visible in LDS
    v16bf PA = make_a(myl + lr * 32, 0, half);      // read back in A-layout
    asm volatile("s_wait_dscnt 0" ::: "memory");    // reads done before next stores

    // ---- ctx accumulation ----
#pragma unroll
    for (int t = 0; t < 4; ++t) acc[t] = wmma_bf16(PA, Bv[t], acc[t]);
  }
#pragma unroll
  for (int t = 0; t < 4; ++t)
#pragma unroll
    for (int r = 0; r < 8; ++r) {
      int sg = qt * 16 + r + half * 8;
      size_t m = (size_t)b * S_ + sg;
      int n = h * 64 + t * 16 + lr;
      ctxb[m * D_ + n] = f2bf(acc[t][r]);
    }
}

// ---------------- out-proj + residual: y = ctx @ Wo.T + x ----------------
__global__ __launch_bounds__(128)
void k_oproj(const u16* __restrict__ ctxb, const u16* __restrict__ wob,
             const float* __restrict__ x, float* __restrict__ y) {
  const int lane = threadIdx.x & 31, lr = lane & 15, half = lane >> 4;
  const int wid = blockIdx.x * (blockDim.x >> 5) + (threadIdx.x >> 5);
  const int nb = wid & 15;
  const int mt = wid >> 4;
  const u16* arow = ctxb + (size_t)(mt * 16 + lr) * D_;
  const u16* wrow[4];
#pragma unroll
  for (int t = 0; t < 4; ++t)
    wrow[t] = wob + (size_t)(nb * 64 + t * 16 + lr) * D_ + half * 16;
  v8f acc[4] = { vzero8(), vzero8(), vzero8(), vzero8() };

  constexpr int NC = D_ / 32;
  v16bf Ap[3];
  v16bf Bp[3][4];
#pragma unroll
  for (int c = 0; c < 2; ++c) {
    Ap[c] = make_a(arow, c * 32, half);
#pragma unroll
    for (int t = 0; t < 4; ++t) Bp[c][t] = ldb16(wrow[t] + c * 32);
  }
#pragma unroll
  for (int c = 0; c < NC; ++c) {
    if (c + 2 < NC) {
      Ap[(c + 2) % 3] = make_a(arow, (c + 2) * 32, half);
#pragma unroll
      for (int t = 0; t < 4; ++t) Bp[(c + 2) % 3][t] = ldb16(wrow[t] + (c + 2) * 32);
    }
#pragma unroll
    for (int t = 0; t < 4; ++t) acc[t] = wmma_bf16(Ap[c % 3], Bp[c % 3][t], acc[t]);
  }
  SCHED_VMEM(20);
#pragma unroll
  for (int c = 0; c < NC - 2; ++c) { SCHED_WMMA(4); SCHED_VMEM(10); }
  SCHED_WMMA(4);
  SCHED_WMMA(4);
  SCHED_FENCE();                      // residual loads/stores stay out of the pipe

#pragma unroll
  for (int t = 0; t < 4; ++t)
#pragma unroll
    for (int r = 0; r < 8; ++r) {
      int m = mt * 16 + r + half * 8;
      int n = nb * 64 + t * 16 + lr;
      size_t idx = (size_t)m * D_ + n;
      y[idx] = acc[t][r] + x[idx];
    }
}

// ---------------- LayerNorm (deterministic tree reduction) ----------------
__global__ __launch_bounds__(256)
void k_ln(const float* __restrict__ y, const float* __restrict__ gamma,
          const float* __restrict__ beta, float* __restrict__ out) {
  __shared__ float ssum[256], ssq[256];
  const int row = blockIdx.x;
  const float* yr = y + (size_t)row * D_;
  float s = 0.f, q = 0.f;
  for (int c = threadIdx.x; c < D_; c += 256) { float v = yr[c]; s += v; q += v * v; }
  ssum[threadIdx.x] = s; ssq[threadIdx.x] = q;
  __syncthreads();
  for (int off = 128; off > 0; off >>= 1) {
    if (threadIdx.x < off) {
      ssum[threadIdx.x] += ssum[threadIdx.x + off];
      ssq[threadIdx.x]  += ssq[threadIdx.x + off];
    }
    __syncthreads();
  }
  float mu  = ssum[0] * (1.0f / D_);
  float var = ssq[0] * (1.0f / D_) - mu * mu;
  float rstd = rsqrtf(var + 1e-5f);
  for (int c = threadIdx.x; c < D_; c += 256)
    out[(size_t)row * D_ + c] = (yr[c] - mu) * rstd * gamma[c] + beta[c];
}

extern "C" void kernel_launch(void* const* d_in, const int* in_sizes, int n_in,
                              void* d_out, int out_size, void* d_ws, size_t ws_size,
                              hipStream_t stream) {
  (void)in_sizes; (void)n_in; (void)out_size; (void)ws_size;
  const float* x     = (const float*)d_in[0];
  const float* wq    = (const float*)d_in[1];
  const float* wk    = (const float*)d_in[2];
  const float* wv    = (const float*)d_in[3];
  const float* wo    = (const float*)d_in[4];
  const float* gamma = (const float*)d_in[5];
  const float* beta  = (const float*)d_in[6];

  char* ws = (char*)d_ws;
  size_t off = 0;
  auto carve = [&](size_t bytes) -> void* {
    void* p = ws + off;
    off += (bytes + 255) & ~(size_t)255;
    return p;
  };
  u16*   xb    = (u16*)  carve((size_t)M_ * D_ * 2);   // x in bf16
  u16*   wqb   = (u16*)  carve((size_t)D_ * D_ * 2);
  u16*   wkb   = (u16*)  carve((size_t)D_ * D_ * 2);
  u16*   wvb   = (u16*)  carve((size_t)D_ * D_ * 2);
  u16*   wob   = (u16*)  carve((size_t)D_ * D_ * 2);
  u16*   qhb   = (u16*)  carve((size_t)M_ * D_ * 2);   // [b,h,s,dh]
  u16*   khb   = (u16*)  carve((size_t)M_ * D_ * 2);   // [b,h,s,dh]
  u16*   vTb   = (u16*)  carve((size_t)M_ * D_ * 2);   // [b,h,dh,s]
  float* colM  = (float*)carve((size_t)B_ * H_ * S_ * 4);
  float* colRZ = (float*)carve((size_t)B_ * H_ * S_ * 4);
  u16*   ctxb  = (u16*)  carve((size_t)M_ * D_ * 2);   // [m, d_model]
  float* ybuf  = (float*)carve((size_t)M_ * D_ * 4);   // pre-LN fp32

  k_to_bf16<<<1024, 256, 0, stream>>>(x,  xb,  M_ * D_);
  k_to_bf16<<<1024, 256, 0, stream>>>(wq, wqb, D_ * D_);
  k_to_bf16<<<1024, 256, 0, stream>>>(wk, wkb, D_ * D_);
  k_to_bf16<<<1024, 256, 0, stream>>>(wv, wvb, D_ * D_);
  k_to_bf16<<<1024, 256, 0, stream>>>(wo, wob, D_ * D_);

  k_proj<<<1024, 128, 0, stream>>>(xb, wqb, qhb, 0);
  k_proj<<<1024, 128, 0, stream>>>(xb, wkb, khb, 0);
  k_proj<<<1024, 128, 0, stream>>>(xb, wvb, vTb, 1);

  k_colstats<<<1024, 128, 0, stream>>>(qhb, khb, colM, colRZ);
  k_ctx<<<1024, 128, 0, stream>>>(qhb, khb, vTb, colM, colRZ, ctxb);
  k_oproj<<<1024, 128, 0, stream>>>(ctxb, wob, x, ybuf);
  k_ln<<<M_, 256, 0, stream>>>(ybuf, gamma, beta, (float*)d_out);
}
